// GNN_VAE_RG_91182155694152
// MI455X (gfx1250) — matrix operand order repaired and checked by reading, not compiled, not run on hardware
//
#include <hip/hip_runtime.h>
#include <hip/hip_bf16.h>

typedef __bf16 v16bf __attribute__((ext_vector_type(16)));
typedef __bf16 v8bf  __attribute__((ext_vector_type(8)));
typedef float  v8f   __attribute__((ext_vector_type(8)));

#define NNODES 262144
#define NEDGES 1048576
#define HD     128
#define BGR    64
#define LDIM   64
#define NN_    4096
#define LDA    136   // padded LDS row stride (bf16 elems): 272B = 17 x 16B -> conflict-free b128

// ---------------- weight f32 -> bf16, transposed: dst[n*K + k] = src[k*N + n] ----------------
__global__ void k_cvt_bf16_t(const float* __restrict__ src, __bf16* __restrict__ dst, int n_total) {
    int idx = blockIdx.x * blockDim.x + threadIdx.x;
    if (idx >= n_total) return;                  // n_total = 128*128
    int k = idx >> 7, n = idx & 127;
    dst[n * HD + k] = (__bf16)src[idx];
}

// ---------------- degree / dinv ----------------
__global__ void k_deg_init(float* deg, int n) {
    int i = blockIdx.x * blockDim.x + threadIdx.x;
    if (i < n) deg[i] = 1.0f;                    // self-loop
}
__global__ void k_deg_scatter(const int* __restrict__ col, float* deg, int e) {
    int i = blockIdx.x * blockDim.x + threadIdx.x;
    if (i < e) atomicAdd(&deg[col[i]], 1.0f);
}
__global__ void k_rsqrt(float* deg, int n) {
    int i = blockIdx.x * blockDim.x + threadIdx.x;
    if (i < n) deg[i] = rsqrtf(deg[i]);          // deg >= 1 always
}

// ---------------- encoder layer 1 (rank-1 xW) : S = x*w1*dinv, Acc = 0 ----------------
__global__ void k_enc1(const float* __restrict__ x, const float* __restrict__ w1,
                       const float* __restrict__ dinv, float* __restrict__ S,
                       float* __restrict__ Acc, int total) {
    int idx = blockIdx.x * blockDim.x + threadIdx.x;
    if (idx >= total) return;
    int node = idx >> 7, f = idx & 127;
    S[idx]   = x[node] * w1[f] * dinv[node];
    Acc[idx] = 0.0f;
}

// ---------------- WMMA GEMM: S = (Hin @ W) * dinv, Acc = 0  (W given transposed bf16) ----------
__global__ __launch_bounds__(256) void k_gemm128_scale(
    const float* __restrict__ Hin, const __bf16* __restrict__ Wt,
    const float* __restrict__ dinv, float* __restrict__ S, float* __restrict__ Acc) {
    __shared__ __bf16 ldsA[128 * LDA];   // activation tile, bf16, padded rows
    __shared__ __bf16 ldsW[128 * LDA];   // W^T  (ldsW[n][k])
    const int tid = threadIdx.x;
    const int rowBase = blockIdx.x * 128;

    // stage W^T as dwords (row remap for padding)
    const uint32_t* wsrc = (const uint32_t*)Wt;          // 8192 dwords
    for (int i = tid; i < HD * HD / 2; i += 256) {
        int n = i >> 6, kq = i & 63;
        ((uint32_t*)(ldsW + n * LDA))[kq] = wsrc[i];
    }
    // stage A with f32->bf16 convert
    const float* asrc = Hin + (size_t)rowBase * HD;
    for (int i = tid; i < 128 * HD; i += 256) {
        int r = i >> 7, k = i & 127;
        ldsA[r * LDA + k] = (__bf16)asrc[i];
    }
    __syncthreads();

    const int wave = tid >> 5, lane = tid & 31;
    const int m = lane & 15, hsel = lane >> 4;
    const int mt = wave >> 1, ct = wave & 1;   // 4 M-subtiles x 2 N-halves of waves
    const int r0 = mt * 32;                    // wave covers rows r0..r0+31
    const int c0 = ct * 64;                    // wave covers cols c0..c0+63

    v8f acc[2][4] = {};
    #pragma unroll
    for (int ks = 0; ks < 4; ++ks) {
        // A fragments: lane holds row (r0+i*16+m), K = ks*32 + hsel*8 + {0..7, 16..23}
        v16bf a[2];
        #pragma unroll
        for (int i = 0; i < 2; ++i) {
            const __bf16* pa = ldsA + (r0 + i * 16 + m) * LDA + ks * 32 + hsel * 8;
            v8bf lo = *(const v8bf*)(pa);
            v8bf hi = *(const v8bf*)(pa + 16);
            a[i] = __builtin_shufflevector(lo, hi, 0,1,2,3,4,5,6,7,8,9,10,11,12,13,14,15);
        }
        #pragma unroll
        for (int j = 0; j < 4; ++j) {
            // B fragment: lane holds col (c0+j*16+m), K = ks*32 + hsel*8 + {0..7, 16..23}
            const __bf16* pb = ldsW + (c0 + j * 16 + m) * LDA + ks * 32 + hsel * 8;
            v8bf lo = *(const v8bf*)(pb);
            v8bf hi = *(const v8bf*)(pb + 16);
            v16bf b = __builtin_shufflevector(lo, hi, 0,1,2,3,4,5,6,7,8,9,10,11,12,13,14,15);
            #pragma unroll
            for (int i = 0; i < 2; ++i) {
                acc[i][j] = __builtin_amdgcn_wmma_f32_16x16x32_bf16(
                    false, a[i], false, b, (short)0, acc[i][j], false, false);
            }
        }
    }
    // epilogue: D layout lanes 0-15 -> M=v, lanes 16-31 -> M=v+8 ; N = lane&15
    #pragma unroll
    for (int i = 0; i < 2; ++i) {
        float dv[8];
        #pragma unroll
        for (int v = 0; v < 8; ++v) dv[v] = dinv[rowBase + r0 + i * 16 + hsel * 8 + v];
        #pragma unroll
        for (int j = 0; j < 4; ++j) {
            #pragma unroll
            for (int v = 0; v < 8; ++v) {
                int row = rowBase + r0 + i * 16 + hsel * 8 + v;
                int idx = row * HD + c0 + j * 16 + m;
                S[idx]   = acc[i][j][v] * dv[v];
                Acc[idx] = 0.0f;
            }
        }
    }
}

// ---------------- edge scatter: Acc[col] += S[row] (4 feats / thread) ----------------
__global__ void k_scatter(const int* __restrict__ row, const int* __restrict__ col,
                          const float* __restrict__ S, float* __restrict__ Acc, int total) {
    int idx = blockIdx.x * blockDim.x + threadIdx.x;
    if (idx >= total) return;
    int e = idx >> 5, q = idx & 31;
    int r = row[e], c = col[e];
    float4 v = ((const float4*)(S + (size_t)r * HD))[q];
    float* dst = Acc + (size_t)c * HD + q * 4;
    atomicAdd(dst + 0, v.x);
    atomicAdd(dst + 1, v.y);
    atomicAdd(dst + 2, v.z);
    atomicAdd(dst + 3, v.w);
}

// ---------------- H = relu((Acc + S_self)*dinv + b) ----------------
__global__ void k_finalize(const float* __restrict__ S, const float* __restrict__ Acc,
                           const float* __restrict__ dinv, const float* __restrict__ bias,
                           float* __restrict__ Hout, int total) {
    int idx = blockIdx.x * blockDim.x + threadIdx.x;
    if (idx >= total) return;
    int node = idx >> 7, f = idx & 127;
    float v = (Acc[idx] + S[idx]) * dinv[node] + bias[f];
    Hout[idx] = v > 0.0f ? v : 0.0f;
}

// ---------------- per-graph mean pool ----------------
__global__ void k_pool(const float* __restrict__ Hin, float* __restrict__ hg) {
    int b = blockIdx.x, f = threadIdx.x;          // 64 blocks x 128 threads
    const float* p = Hin + (size_t)b * NN_ * HD + f;
    float s = 0.0f;
    for (int j = 0; j < NN_; ++j) s += p[(size_t)j * HD];
    hg[b * HD + f] = s * (1.0f / (float)NN_);
}

// ---------------- mu / logvar / z ----------------
__global__ void k_latent(const float* __restrict__ hg,
                         const float* __restrict__ mu_w, const float* __restrict__ mu_b,
                         const float* __restrict__ lv_w, const float* __restrict__ lv_b,
                         const float* __restrict__ eps,
                         float* __restrict__ out_mu, float* __restrict__ out_lv,
                         float* __restrict__ out_z) {
    int idx = blockIdx.x * blockDim.x + threadIdx.x;
    if (idx >= BGR * LDIM) return;
    int b = idx >> 6, l = idx & 63;
    const float* h = hg + b * HD;
    float mu = mu_b[l], lv = lv_b[l];
    for (int k = 0; k < HD; ++k) {
        float hv = h[k];
        mu += hv * mu_w[k * LDIM + l];
        lv += hv * lv_w[k * LDIM + l];
    }
    out_mu[idx] = mu;
    out_lv[idx] = lv;
    out_z[idx]  = mu + expf(0.5f * lv) * eps[idx];
}

// ---------------- zn = z @ zn_w + zn_b ----------------
__global__ void k_zn(const float* __restrict__ z, const float* __restrict__ zn_w,
                     const float* __restrict__ zn_b, float* __restrict__ zn) {
    int idx = blockIdx.x * blockDim.x + threadIdx.x;
    if (idx >= BGR * HD) return;
    int b = idx >> 7, h = idx & 127;
    float s = zn_b[h];
    for (int l = 0; l < LDIM; ++l) s += z[b * LDIM + l] * zn_w[l * HD + h];
    zn[idx] = s;
}

// ---------------- broadcast zn to nodes ----------------
__global__ void k_bcast(const float* __restrict__ zn, float* __restrict__ Hout, int total) {
    int idx = blockIdx.x * blockDim.x + threadIdx.x;
    if (idx >= total) return;
    int node = idx >> 7, f = idx & 127;
    Hout[idx] = zn[(node >> 12) * HD + f];        // batch = node / 4096
}

// ---------------- recon = tanh(H @ out_w + out_b) ----------------
__global__ void k_out(const float* __restrict__ Hin, const float* __restrict__ ow,
                      const float* __restrict__ ob, float* __restrict__ recon, int n) {
    int node = blockIdx.x * blockDim.x + threadIdx.x;
    if (node >= n) return;
    const float4* p = (const float4*)(Hin + (size_t)node * HD);
    const float4* w = (const float4*)ow;
    float s = 0.0f;
    #pragma unroll 8
    for (int q = 0; q < 32; ++q) {
        float4 a = p[q], b = w[q];
        s += a.x * b.x + a.y * b.y + a.z * b.z + a.w * b.w;
    }
    recon[node] = tanhf(s + ob[0]);
}

extern "C" void kernel_launch(void* const* d_in, const int* in_sizes, int n_in,
                              void* d_out, int out_size, void* d_ws, size_t ws_size,
                              hipStream_t stream) {
    const float* x      = (const float*)d_in[0];
    const float* eps    = (const float*)d_in[1];
    const float* enc_w1 = (const float*)d_in[2];
    const float* enc_b1 = (const float*)d_in[3];
    const float* enc_w2 = (const float*)d_in[4];
    const float* enc_b2 = (const float*)d_in[5];
    const float* mu_w   = (const float*)d_in[6];
    const float* mu_b   = (const float*)d_in[7];
    const float* lv_w   = (const float*)d_in[8];
    const float* lv_b   = (const float*)d_in[9];
    const float* zn_w   = (const float*)d_in[10];
    const float* zn_b   = (const float*)d_in[11];
    const float* dec_w1 = (const float*)d_in[12];
    const float* dec_b1 = (const float*)d_in[13];
    const float* dec_w2 = (const float*)d_in[14];
    const float* dec_b2 = (const float*)d_in[15];
    const float* out_w  = (const float*)d_in[16];
    const float* out_b  = (const float*)d_in[17];
    const int*   eidx   = (const int*)d_in[18];
    const int* row = eidx;                 // edge_index[0]
    const int* col = eidx + NEDGES;        // edge_index[1]

    // workspace layout
    char* p = (char*)d_ws;
    float* dinv = (float*)p;  p += (size_t)NNODES * 4;
    float* bufS = (float*)p;  p += (size_t)NNODES * HD * 4;
    float* bufA = (float*)p;  p += (size_t)NNODES * HD * 4;
    float* bufH = (float*)p;  p += (size_t)NNODES * HD * 4;
    float* hg   = (float*)p;  p += (size_t)BGR * HD * 4;
    float* znb  = (float*)p;  p += (size_t)BGR * HD * 4;
    __bf16* w2t  = (__bf16*)p; p += (size_t)HD * HD * 2;
    __bf16* dw1t = (__bf16*)p; p += (size_t)HD * HD * 2;
    __bf16* dw2t = (__bf16*)p; p += (size_t)HD * HD * 2;

    float* out_recon = (float*)d_out;
    float* out_mu = out_recon + NNODES;
    float* out_lv = out_mu + BGR * LDIM;
    float* out_z  = out_lv + BGR * LDIM;

    const int T = 256;
    const int totF   = NNODES * HD;           // 33.5M
    const int gF     = totF / T;              // 131072
    const int totSc  = NEDGES * 32;           // 33.5M
    const int gSc    = totSc / T;
    const int gGemm  = NNODES / 128;          // 2048

    // weights -> bf16 transposed
    k_cvt_bf16_t<<<(HD*HD + T - 1) / T, T, 0, stream>>>(enc_w2, w2t, HD * HD);
    k_cvt_bf16_t<<<(HD*HD + T - 1) / T, T, 0, stream>>>(dec_w1, dw1t, HD * HD);
    k_cvt_bf16_t<<<(HD*HD + T - 1) / T, T, 0, stream>>>(dec_w2, dw2t, HD * HD);

    // symmetric GCN norm via dinv
    k_deg_init<<<NNODES / T, T, 0, stream>>>(dinv, NNODES);
    k_deg_scatter<<<NEDGES / T, T, 0, stream>>>(col, dinv, NEDGES);
    k_rsqrt<<<NNODES / T, T, 0, stream>>>(dinv, NNODES);

    // encoder layer 1 (rank-1 GEMM fused)
    k_enc1<<<gF, T, 0, stream>>>(x, enc_w1, dinv, bufS, bufA, totF);
    k_scatter<<<gSc, T, 0, stream>>>(row, col, bufS, bufA, totSc);
    k_finalize<<<gF, T, 0, stream>>>(bufS, bufA, dinv, enc_b1, bufH, totF);

    // encoder layer 2 (WMMA)
    k_gemm128_scale<<<gGemm, T, 0, stream>>>(bufH, w2t, dinv, bufS, bufA);
    k_scatter<<<gSc, T, 0, stream>>>(row, col, bufS, bufA, totSc);
    k_finalize<<<gF, T, 0, stream>>>(bufS, bufA, dinv, enc_b2, bufH, totF);

    // pool + latent
    k_pool<<<BGR, HD, 0, stream>>>(bufH, hg);
    k_latent<<<(BGR*LDIM + T - 1) / T, T, 0, stream>>>(hg, mu_w, mu_b, lv_w, lv_b, eps,
                                                       out_mu, out_lv, out_z);
    k_zn<<<(BGR*HD + T - 1) / T, T, 0, stream>>>(out_z, zn_w, zn_b, znb);
    k_bcast<<<gF, T, 0, stream>>>(znb, bufH, totF);

    // decoder layer 1 (WMMA)
    k_gemm128_scale<<<gGemm, T, 0, stream>>>(bufH, dw1t, dinv, bufS, bufA);
    k_scatter<<<gSc, T, 0, stream>>>(row, col, bufS, bufA, totSc);
    k_finalize<<<gF, T, 0, stream>>>(bufS, bufA, dinv, dec_b1, bufH, totF);

    // decoder layer 2 (WMMA)
    k_gemm128_scale<<<gGemm, T, 0, stream>>>(bufH, dw2t, dinv, bufS, bufA);
    k_scatter<<<gSc, T, 0, stream>>>(row, col, bufS, bufA, totSc);
    k_finalize<<<gF, T, 0, stream>>>(bufS, bufA, dinv, dec_b2, bufH, totF);

    // output head
    k_out<<<NNODES / T, T, 0, stream>>>(bufH, out_w, out_b, out_recon, NNODES);
}
